// MultiHeadAttention_49331994362543
// MI455X (gfx1250) — matrix operand order repaired and checked by reading, not compiled, not run on hardware
//
#include <hip/hip_runtime.h>

// ---------------------------------------------------------------------------
// MI455X / gfx1250 GQA attention layer, bf16 WMMA everywhere.
//   B=1, T=2048, D=4096, 32 q heads / 8 kv heads (G=4), head dim 128.
// ---------------------------------------------------------------------------

typedef __bf16 bf16;
typedef __attribute__((ext_vector_type(16))) bf16  v16bf;
typedef __attribute__((ext_vector_type(8)))  float v8f;

#define T_SEQ   2048
#define D_MODEL 4096
#define N_QH    32
#define N_KVH   8
#define DH      128

__device__ __forceinline__ bf16 tobf(float x) { return (bf16)x; }

__device__ __forceinline__ v8f wmma_bf16(v16bf a, v16bf b, v8f c) {
  // D(f32 16x16) = A(bf16 16x32) * B(bf16 32x16) + C
  return __builtin_amdgcn_wmma_f32_16x16x32_bf16(false, a, false, b,
                                                 (short)0, c, false, false);
}

// A fragment (16x32 bf16) from row-major storage with leading dim `lda`.
// Layout: lanes 0-15 row M=lane hold K={0..15}; lanes 16-31 hold
// the same rows with K={16..31} (matches B fragments, k = h*16 + e).
__device__ __forceinline__ v16bf load_a_frag(const bf16* p, int lda, int lane) {
  const int r = lane & 15, h = lane >> 4;
  const bf16* row = p + (size_t)r * lda;
  v16bf out;
#pragma unroll
  for (int e = 0; e < 8; ++e) out[e] = row[h * 16 + e];
#pragma unroll
  for (int e = 0; e < 8; ++e) out[8 + e] = row[h * 16 + 8 + e];
  return out;
}

// B fragment helper: p must point at 16 contiguous bf16 = B[k..k+15][n] for
// this lane's column n and half h (k = h*16 + e).
__device__ __forceinline__ v16bf load_row16(const bf16* p) {
  v16bf out;
#pragma unroll
  for (int e = 0; e < 16; ++e) out[e] = p[e];
  return out;
}

struct alignas(16) bf16x16 { bf16 e[16]; };

// ---------------------------------------------------------------------------
// GEMM: C[M,N](f32) = A[M,K](f32, row-major) * B[K,N](f32, row-major),
// computed in bf16 WMMA with f32 accumulate. M,N % 128 == 0, K % 64 == 0.
// Register-double-buffered: next tile's 16 global_load_b128 are issued right
// after the barrier and retire while the current tile's 16 WMMAs run.
// ---------------------------------------------------------------------------
#define BM 128
#define BN 128
#define BK 64

__global__ __launch_bounds__(256, 1)
void gemm_f32_bf16wmma(const float* __restrict__ A, const float* __restrict__ B,
                       float* __restrict__ C, int M, int N, int K) {
  __shared__ bf16 As[BM][BK + 8];
  __shared__ bf16 Bt[BN][BK + 8];   // B tile stored transposed: Bt[n][k]

  const int tid  = threadIdx.x;
  const int lane = tid & 31, wave = tid >> 5;
  const int wm = wave >> 1, wn = wave & 1;   // 4x2 wave grid -> 32x64 per wave
  const int row0 = blockIdx.y * BM, col0 = blockIdx.x * BN;
  const int r = lane & 15, h = lane >> 4;

  // Staging registers: A tile 128x64, B tile 64x128, 8 float4 each per thread.
  float4 ra[8], rb[8];

  auto load_tiles = [&](int k0) {
#pragma unroll
    for (int it = 0; it < 8; ++it) {
      const int i = tid * 4 + it * 1024;           // uniform trip count
      ra[it] = *(const float4*)(A + (size_t)(row0 + (i >> 6)) * K + k0 + (i & 63));
    }
#pragma unroll
    for (int it = 0; it < 8; ++it) {
      const int i = tid * 4 + it * 1024;
      rb[it] = *(const float4*)(B + (size_t)(k0 + (i >> 7)) * N + col0 + (i & 127));
    }
  };
  auto store_tiles = [&]() {
#pragma unroll
    for (int it = 0; it < 8; ++it) {
      const int i = tid * 4 + it * 1024;
      const int rr = i >> 6, cc = i & 63;
      As[rr][cc + 0] = tobf(ra[it].x);
      As[rr][cc + 1] = tobf(ra[it].y);
      As[rr][cc + 2] = tobf(ra[it].z);
      As[rr][cc + 3] = tobf(ra[it].w);
    }
#pragma unroll
    for (int it = 0; it < 8; ++it) {
      const int i = tid * 4 + it * 1024;
      const int rr = i >> 7, cc = i & 127;
      Bt[cc + 0][rr] = tobf(rb[it].x);
      Bt[cc + 1][rr] = tobf(rb[it].y);
      Bt[cc + 2][rr] = tobf(rb[it].z);
      Bt[cc + 3][rr] = tobf(rb[it].w);
    }
  };

  v8f acc[2][4];
  const v8f vzero = {0.f, 0.f, 0.f, 0.f, 0.f, 0.f, 0.f, 0.f};
#pragma unroll
  for (int mi = 0; mi < 2; ++mi)
#pragma unroll
    for (int ni = 0; ni < 4; ++ni) acc[mi][ni] = vzero;

  load_tiles(0);
  for (int k0 = 0; k0 < K; k0 += BK) {
    store_tiles();                       // regs (prev loads) -> LDS as bf16
    __syncthreads();
    if (k0 + BK < K) load_tiles(k0 + BK);          // overlap with WMMAs below
    if (k0 + 2 * BK < K) {                          // warm L2 two tiles ahead
      __builtin_prefetch(
          A + (size_t)(row0 + (tid & 127)) * K + k0 + 2 * BK + (tid >> 7) * 32,
          0, 3);
      __builtin_prefetch(
          B + (size_t)(k0 + 2 * BK + (tid & 63)) * N + col0 + (tid >> 6) * 32,
          0, 3);
    }
#pragma unroll
    for (int ks = 0; ks < 2; ++ks) {                // two WMMA K-steps per tile
      v16bf af[2], bfr[4];
#pragma unroll
      for (int mi = 0; mi < 2; ++mi)
        af[mi] = load_a_frag(&As[wm * 32 + mi * 16][ks * 32], BK + 8, lane);
#pragma unroll
      for (int ni = 0; ni < 4; ++ni)
        bfr[ni] = load_row16(&Bt[wn * 64 + ni * 16 + r][ks * 32 + h * 16]);
#pragma unroll
      for (int mi = 0; mi < 2; ++mi)
#pragma unroll
        for (int ni = 0; ni < 4; ++ni)
          acc[mi][ni] = wmma_bf16(af[mi], bfr[ni], acc[mi][ni]);
    }
    __syncthreads();
  }

  // Epilogue: f32 C/D layout — VGPR v holds row v (lanes 0-15) / v+8 (16-31).
#pragma unroll
  for (int mi = 0; mi < 2; ++mi)
#pragma unroll
    for (int ni = 0; ni < 4; ++ni)
#pragma unroll
      for (int v = 0; v < 8; ++v) {
        const int rr = row0 + wm * 32 + mi * 16 + v + 8 * h;
        const int cc = col0 + wn * 64 + ni * 16 + r;
        C[(size_t)rr * N + cc] = acc[mi][ni][v];
      }
}

// ---------------------------------------------------------------------------
// RoPE + head packing. One block per timestep t.
//  - q_heads -> rotary -> bf16 [hq][T][128]
//  - k_heads -> rotary -> fp32 kv-cache (c<128) + bf16 [hk][T][128]
//  - v_heads -> fp32 kv-cache (c>=128) + bf16 [hk][T][128]
// ---------------------------------------------------------------------------
__global__ __launch_bounds__(256, 1)
void rope_pack(const float* __restrict__ qh, const float* __restrict__ kh,
               const float* __restrict__ vh, bf16* __restrict__ qb,
               bf16* __restrict__ kb, bf16* __restrict__ vb,
               float* __restrict__ kvout) {
  const int t = blockIdx.x;
  const int tid = threadIdx.x;
  const float tt = (float)t;
  const float kFreq = -0.14391156816f;  // -2*ln(10000)/128

  for (int i = tid; i < N_QH * DH; i += 256) {
    const int hd = i >> 7, d = i & 127, f = d & 63;
    const float ang = tt * __expf(kFreq * (float)f);
    const float cs = __cosf(ang), sn = __sinf(ang);
    const size_t base = (size_t)t * (N_QH * DH) + (size_t)hd * DH;
    const float x = qh[base + d];
    const float other = (d < 64) ? -qh[base + d + 64] : qh[base + d - 64];
    qb[((size_t)hd * T_SEQ + t) * DH + d] = tobf(x * cs + other * sn);
  }
  for (int i = tid; i < N_KVH * DH; i += 256) {
    const int hd = i >> 7, d = i & 127, f = d & 63;
    const float ang = tt * __expf(kFreq * (float)f);
    const float cs = __cosf(ang), sn = __sinf(ang);
    const size_t base = (size_t)t * (N_KVH * DH) + (size_t)hd * DH;
    const float x = kh[base + d];
    const float other = (d < 64) ? -kh[base + d + 64] : kh[base + d - 64];
    const float val = x * cs + other * sn;
    kvout[((size_t)t * N_KVH + hd) * 256 + d] = val;
    kb[((size_t)hd * T_SEQ + t) * DH + d] = tobf(val);
  }
  for (int i = tid; i < N_KVH * DH; i += 256) {
    const int hd = i >> 7, d = i & 127;
    const float val = vh[(size_t)t * (N_KVH * DH) + i];
    kvout[((size_t)t * N_KVH + hd) * 256 + 128 + d] = val;
    vb[((size_t)hd * T_SEQ + t) * DH + d] = tobf(val);
  }
}

// ---------------------------------------------------------------------------
// Causal flash attention, GQA (q head hq uses kv head hq%8).
// grid = (32 heads, T/128); 8 waves x 16 query rows each.
// All waves run a uniform key-block loop to the block's causal top and
// cooperatively stage K (GLOBAL_LOAD_ASYNC_TO_LDS_B128) and V (transposed,
// manual) into LDS; per-wave compute is guarded by its own causal bound
// (wave-uniform branch, EXEC stays all-ones around every WMMA).
// ---------------------------------------------------------------------------
__global__ __launch_bounds__(256, 1)
void attn_flash(const bf16* __restrict__ qb, const bf16* __restrict__ kb,
                const bf16* __restrict__ vb, float* __restrict__ ao) {
  __shared__ bf16 Ks[32][136];     // K block, row-major, padded pitch
  __shared__ bf16 Vt[128][40];     // V block transposed: Vt[dim][key]
  __shared__ bf16 Ps[8][16][40];   // per-wave 16x32 P scratch (+pad)

  const int tid  = threadIdx.x;
  const int lane = tid & 31, wave = tid >> 5;
  const int r = lane & 15, h = lane >> 4;
  const int hq = blockIdx.x, hk = hq & (N_KVH - 1);
  const int qBase = blockIdx.y * 128 + wave * 16;
  const int qTop  = blockIdx.y * 128 + 127;        // block-uniform loop bound
  const float scale = 0.08838834764831845f;        // 1/sqrt(128)

  // Q tile (16x128) as four A fragments, kept resident.
  const bf16* qptr = qb + ((size_t)hq * T_SEQ + qBase) * DH;
  v16bf aq[4];
#pragma unroll
  for (int kk = 0; kk < 4; ++kk) aq[kk] = load_a_frag(qptr + kk * 32, DH, lane);

  float m[8], l[8];
  v8f o[8];
  const v8f vzero = {0.f, 0.f, 0.f, 0.f, 0.f, 0.f, 0.f, 0.f};
#pragma unroll
  for (int v = 0; v < 8; ++v) { m[v] = -__builtin_inff(); l[v] = 0.f; }
#pragma unroll
  for (int ni = 0; ni < 8; ++ni) o[ni] = vzero;

  // Cooperative staging coordinates (uniform across the loop).
  const int krow0 = tid >> 4;          // K rows: krow0 and krow0+16
  const int kseg  = tid & 15;          // 16B segment within a 256B K row
  const int vrow  = tid >> 3;          // V row 0..31
  const int vcol0 = (tid & 7) * 16;    // V 16-col group

  for (int c0 = 0; c0 <= qTop; c0 += 32) {   // uniform key-block loop
    __syncthreads();   // previous iteration's LDS readers are done

    // --- K block 32x128 bf16: async global->LDS copy (2 x b128/thread) ---
#pragma unroll
    for (int it = 0; it < 2; ++it) {
      const int row = krow0 + it * 16;
      const unsigned lds = (unsigned)(uintptr_t)(&Ks[row][kseg * 8]);
      const unsigned long long src = (unsigned long long)(uintptr_t)(
          kb + ((size_t)hk * T_SEQ + c0 + row) * DH + kseg * 8);
      asm volatile("global_load_async_to_lds_b128 %0, %1, off"
                   :: "v"(lds), "v"(src) : "memory");
    }
    // --- V block 32x128 bf16: staged transposed (needs relayout) ---
    {
      const bf16x16 raw = *(const bf16x16*)(
          vb + ((size_t)hk * T_SEQ + c0 + vrow) * DH + vcol0);
#pragma unroll
      for (int e = 0; e < 16; ++e) Vt[vcol0 + e][vrow] = raw.e[e];
    }
    asm volatile("s_wait_asynccnt 0x0" ::: "memory");
    __syncthreads();   // staged K/V visible to all waves

    if (c0 <= qBase + 15) {            // wave-uniform causal guard
      // S = Q * K^T : two 16x16 tiles covering keys [c0, c0+32)
      v8f s2[2] = {vzero, vzero};
#pragma unroll
      for (int t2 = 0; t2 < 2; ++t2) {
#pragma unroll
        for (int kk = 0; kk < 4; ++kk) {
          // B[d][n] = K[c0 + t2*16 + n][kk*32 + d]; 16 contiguous bf16/lane.
          const v16bf bk = load_row16(&Ks[t2 * 16 + r][kk * 32 + h * 16]);
          s2[t2] = wmma_bf16(aq[kk], bk, s2[t2]);
        }
      }

      // Online softmax (rows live in VGPR index v, cols in lanes).
      float mnew[8];
#pragma unroll
      for (int v = 0; v < 8; ++v) {
        const int rowg = qBase + v + 8 * h;
        float a0 = s2[0][v] * scale;
        float a1 = s2[1][v] * scale;
        if (c0 + r > rowg)      a0 = -__builtin_inff();
        if (c0 + 16 + r > rowg) a1 = -__builtin_inff();
        s2[0][v] = a0;
        s2[1][v] = a1;
        float tmax = fmaxf(a0, a1);
#pragma unroll
        for (int off = 1; off < 16; off <<= 1)
          tmax = fmaxf(tmax, __shfl_xor(tmax, off, 32));
        mnew[v] = fmaxf(m[v], tmax);
      }
#pragma unroll
      for (int v = 0; v < 8; ++v) {
        const float corr = __expf(m[v] - mnew[v]);
        m[v] = mnew[v];
        l[v] = l[v] * corr;
#pragma unroll
        for (int ni = 0; ni < 8; ++ni) o[ni][v] = o[ni][v] * corr;
        float rs = 0.f;
#pragma unroll
        for (int t2 = 0; t2 < 2; ++t2) {
          const float p = __expf(s2[t2][v] - mnew[v]);
          rs += p;
          Ps[wave][v + 8 * h][t2 * 16 + r] = tobf(p);  // C->A relayout via LDS
        }
#pragma unroll
        for (int off = 1; off < 16; off <<= 1) rs += __shfl_xor(rs, off, 32);
        l[v] += rs;
      }

      // O += P * V : P as A fragment (wave-private LDS; DS ops in-order).
      const v16bf pa = load_a_frag(&Ps[wave][0][0], 40, lane);
#pragma unroll
      for (int ni = 0; ni < 8; ++ni) {
        // B[k][n] = V[c0+k][ni*16+n] = Vt[ni*16+n][k]; contiguous per lane.
        const v16bf bv = load_row16(&Vt[ni * 16 + r][h * 16]);
        o[ni] = wmma_bf16(pa, bv, o[ni]);
      }
    }
  }

  // Normalize and write [t][hq*128 + d] (matches reference BtGHV flattening).
#pragma unroll
  for (int v = 0; v < 8; ++v) {
    const float invl = 1.0f / l[v];
    const int rowg = qBase + v + 8 * h;
#pragma unroll
    for (int ni = 0; ni < 8; ++ni)
      ao[(size_t)rowg * (N_QH * DH) + (size_t)hq * DH + ni * 16 + r] =
          o[ni][v] * invl;
  }
}

// ---------------------------------------------------------------------------
// Launch: qkv projections -> rope/pack -> flash attention -> output proj.
// Workspace (~109 MB): qh(32M) kh(8M) vh(8M) ao(32M) qb(16M) kb(4M) vb(4M).
// ---------------------------------------------------------------------------
extern "C" void kernel_launch(void* const* d_in, const int* in_sizes, int n_in,
                              void* d_out, int out_size, void* d_ws, size_t ws_size,
                              hipStream_t stream) {
  (void)in_sizes; (void)n_in; (void)out_size; (void)ws_size;
  const float* q  = (const float*)d_in[0];
  const float* k  = (const float*)d_in[1];
  const float* v  = (const float*)d_in[2];
  const float* wq = (const float*)d_in[3];
  const float* wk = (const float*)d_in[4];
  const float* wv = (const float*)d_in[5];
  const float* wo = (const float*)d_in[6];

  float* out   = (float*)d_out;                       // [T, D]
  float* kvout = out + (size_t)T_SEQ * D_MODEL;       // [T, 8, 256]

  float* qh = (float*)d_ws;                           // [T, 4096]
  float* kh = qh + (size_t)T_SEQ * D_MODEL;           // [T, 1024]
  float* vh = kh + (size_t)T_SEQ * N_KVH * DH;        // [T, 1024]
  float* ao = vh + (size_t)T_SEQ * N_KVH * DH;        // [T, 4096]
  bf16* qbuf = (bf16*)(ao + (size_t)T_SEQ * D_MODEL); // [32][T][128]
  bf16* kbuf = qbuf + (size_t)N_QH  * T_SEQ * DH;     // [8][T][128]
  bf16* vbuf = kbuf + (size_t)N_KVH * T_SEQ * DH;     // [8][T][128]

  const dim3 blk(256);
  gemm_f32_bf16wmma<<<dim3(D_MODEL / BN, T_SEQ / BM), blk, 0, stream>>>(
      q, wq, qh, T_SEQ, D_MODEL, D_MODEL);
  gemm_f32_bf16wmma<<<dim3((N_KVH * DH) / BN, T_SEQ / BM), blk, 0, stream>>>(
      k, wk, kh, T_SEQ, N_KVH * DH, D_MODEL);
  gemm_f32_bf16wmma<<<dim3((N_KVH * DH) / BN, T_SEQ / BM), blk, 0, stream>>>(
      v, wv, vh, T_SEQ, N_KVH * DH, D_MODEL);
  rope_pack<<<dim3(T_SEQ), blk, 0, stream>>>(qh, kh, vh, qbuf, kbuf, vbuf, kvout);
  attn_flash<<<dim3(N_QH, T_SEQ / 128), blk, 0, stream>>>(qbuf, kbuf, vbuf, ao);
  gemm_f32_bf16wmma<<<dim3(D_MODEL / BN, T_SEQ / BM), blk, 0, stream>>>(
      ao, wo, out, T_SEQ, D_MODEL, D_MODEL);
}
